// NPTransitionPrior_14353780703781
// MI455X (gfx1250) — compile-verified, matrix operand order checked by reading, not run on hardware
//
#include <hip/hip_runtime.h>
#include <hip/hip_fp16.h>
#include <math.h>

typedef __attribute__((ext_vector_type(16))) _Float16 v16h;
typedef __attribute__((ext_vector_type(8)))  float    v8f;

#define NLAGS 2
#define DCH  64          // D: number of channels / feature dim
#define HID  64
#define FIN  129         // LAGS*D + 1
#define SLOPE 0.01f
#define SEQB 16          // batch
#define SEQLEN 1024      // T - LAGS
#define NTOT (SEQB*SEQLEN)   // 16384 samples per channel
#define ROWS_PER_BLOCK 128
#define NWAVES 8
#define NTILES (NTOT/ROWS_PER_BLOCK)   // 128

// Build a 16x32 f16 A/B operand fragment per the CDNA5 WMMA VGPR layout:
// lane half = lane>>4 selects K groups [kb..kb+7] (elems 0..7) and
// [kb+16..kb+23] (elems 8..15).
__device__ inline v16h frag_from_f16(const _Float16* __restrict__ p, int kb) {
    v16h v;
#pragma unroll
    for (int i = 0; i < 8; ++i) {
        v[i]     = p[kb + i];
        v[8 + i] = p[kb + 16 + i];
    }
    return v;
}

__device__ inline v16h frag_from_f32(const float* __restrict__ p, int kb) {
    v16h v;
#pragma unroll
    for (int i = 0; i < 8; ++i) {
        v[i]     = (_Float16)p[kb + i];
        v[8 + i] = (_Float16)p[kb + 16 + i];
    }
    return v;
}

__global__ void zero_lad_kernel(float* __restrict__ lad, int n) {
    int i = blockIdx.x * blockDim.x + threadIdx.x;
    if (i < n) lad[i] = 0.0f;
}

__global__ __launch_bounds__(256)
void np_prior_kernel(const float* __restrict__ x,
                     const float* __restrict__ W1,
                     const float* __restrict__ b1,
                     const float* __restrict__ W2,
                     const float* __restrict__ b2,
                     const float* __restrict__ W3,
                     const float* __restrict__ b3,
                     float* __restrict__ out_res,
                     float* __restrict__ out_lad) {
    // LDS: per-channel weights in f16 (B operands, K-major per column),
    // plus wave-private a1/t1 staging for the GEMM1 -> GEMM2 handoff.
    __shared__ _Float16 W1s[HID * 128];          // [h][k], k = 0..127 (drop last col)
    __shared__ _Float16 W2s[HID * HID];          // [g][k]
    __shared__ float w1l_s[HID], b1_s[HID], b2_s[HID], w3_s[HID];
    __shared__ float b3_s;
    __shared__ _Float16 a1_s[NWAVES * 16 * HID]; // [wave][m][h]
    __shared__ _Float16 t1_s[NWAVES * 16 * HID];

    const int l    = blockIdx.x / NTILES;   // channel
    const int tile = blockIdx.x % NTILES;   // 128-row tile of samples
    const int tid  = threadIdx.x;

    // ---- cooperative weight load (fp32 -> f16 for WMMA B operands) ----
    for (int i = tid; i < HID * 128; i += 256) {
        int h = i >> 7, k = i & 127;
        W1s[i] = (_Float16)W1[((size_t)l * HID + h) * FIN + k];
    }
    for (int i = tid; i < HID * HID; i += 256) {
        int g = i >> 6, k = i & 63;
        W2s[i] = (_Float16)W2[((size_t)l * HID + g) * HID + k];
    }
    if (tid < HID) {
        w1l_s[tid] = W1[((size_t)l * HID + tid) * FIN + 128];
        b1_s[tid]  = b1[l * HID + tid];
        b2_s[tid]  = b2[l * HID + tid];
        w3_s[tid]  = W3[l * HID + tid];
    }
    if (tid == 0) b3_s = b3[l];
    __syncthreads();

    const int wave = tid >> 5;
    const int lane = tid & 31;
    const int half = lane >> 4;    // 0: lanes 0-15, 1: lanes 16-31
    const int lx   = lane & 15;

    const int row0 = tile * ROWS_PER_BLOCK + wave * 16;  // first sample n of this wave
    const int bidx = row0 >> 10;                          // 128 | 1024, so no b-crossing
    const int t0   = row0 & (SEQLEN - 1);
    const float* xb = x + (size_t)bidx * 1026 * DCH;

    // ---- A operand for GEMM1: x_lags rows are 128 contiguous floats at x[b,t,0]
    v16h A1[4];
    {
        const float* rowp = xb + (size_t)(t0 + lx) * DCH;  // row M = lx
#pragma unroll
        for (int kt = 0; kt < 4; ++kt)
            A1[kt] = frag_from_f32(rowp, kt * 32 + half * 8);
    }

    // per-lane x_t values for the 8 C/D rows this lane holds (M = r + 8*half)
    float xt[8];
#pragma unroll
    for (int r = 0; r < 8; ++r) {
        int m = r + 8 * half;
        xt[r] = xb[(size_t)(t0 + m + 2) * DCH + l];
    }

    // ---- GEMM1: z1 = x_lags * W1_lags^T, fused epilogue -> a1, t1 in LDS ----
#pragma unroll
    for (int ct = 0; ct < 4; ++ct) {
        v8f acc = {};
#pragma unroll
        for (int kt = 0; kt < 4; ++kt) {
            v16h bm = frag_from_f16(&W1s[(ct * 16 + lx) * 128], kt * 32 + half * 8);
            acc = __builtin_amdgcn_wmma_f32_16x16x32_f16(
                false, A1[kt], false, bm, (short)0, acc, false, false);
        }
        const int h   = ct * 16 + lx;
        const float wl = w1l_s[h];
        const float bb = b1_s[h];
#pragma unroll
        for (int r = 0; r < 8; ++r) {
            int m = r + 8 * half;
            float z  = acc[r] + xt[r] * wl + bb;
            float av = (z > 0.0f) ? z : SLOPE * z;
            float tv = ((z > 0.0f) ? 1.0f : SLOPE) * wl;
            a1_s[(wave * 16 + m) * HID + h] = (_Float16)av;
            t1_s[(wave * 16 + m) * HID + h] = (_Float16)tv;
        }
    }
    __syncthreads();

    // ---- A operands for GEMM2 from staged a1 / t1 ----
    v16h A2[2], T2[2];
    {
        const _Float16* ap = &a1_s[(wave * 16 + lx) * HID];
        const _Float16* tp = &t1_s[(wave * 16 + lx) * HID];
#pragma unroll
        for (int kt = 0; kt < 2; ++kt) {
            A2[kt] = frag_from_f16(ap, kt * 32 + half * 8);
            T2[kt] = frag_from_f16(tp, kt * 32 + half * 8);
        }
    }

    // ---- GEMM2 (shared B operand for both products) + w3 contraction ----
    float so[8], sd[8];
#pragma unroll
    for (int r = 0; r < 8; ++r) { so[r] = 0.0f; sd[r] = 0.0f; }

#pragma unroll
    for (int ct = 0; ct < 4; ++ct) {
        v8f z2 = {}, u2 = {};
#pragma unroll
        for (int kt = 0; kt < 2; ++kt) {
            v16h bm = frag_from_f16(&W2s[(ct * 16 + lx) * HID], kt * 32 + half * 8);
            z2 = __builtin_amdgcn_wmma_f32_16x16x32_f16(
                false, A2[kt], false, bm, (short)0, z2, false, false);
            u2 = __builtin_amdgcn_wmma_f32_16x16x32_f16(
                false, T2[kt], false, bm, (short)0, u2, false, false);
        }
        const int g    = ct * 16 + lx;
        const float bb = b2_s[g];
        const float w3v = w3_s[g];
#pragma unroll
        for (int r = 0; r < 8; ++r) {
            float z  = z2[r] + bb;
            float a2 = (z > 0.0f) ? z : SLOPE * z;
            float t2 = ((z > 0.0f) ? 1.0f : SLOPE) * u2[r];
            so[r] += a2 * w3v;
            sd[r] += t2 * w3v;
        }
    }

    // ---- reduce across the 16 lanes holding columns g (wave32 shuffles) ----
#pragma unroll
    for (int mask = 1; mask < 16; mask <<= 1) {
#pragma unroll
        for (int r = 0; r < 8; ++r) {
            so[r] += __shfl_xor(so[r], mask, 32);
            sd[r] += __shfl_xor(sd[r], mask, 32);
        }
    }

    if (lx == 0) {
        const float b3v = b3_s;
#pragma unroll
        for (int r = 0; r < 8; ++r) {
            int m = r + 8 * half;
            int n = row0 + m;
            out_res[(size_t)n * DCH + l] = so[r] + b3v;           // residuals[b,t,l]
            atomicAdd(&out_lad[n], logf(fabsf(sd[r])));            // sum_l log|dJ|
        }
    }
}

extern "C" void kernel_launch(void* const* d_in, const int* in_sizes, int n_in,
                              void* d_out, int out_size, void* d_ws, size_t ws_size,
                              hipStream_t stream) {
    (void)in_sizes; (void)n_in; (void)out_size; (void)d_ws; (void)ws_size;
    const float* x  = (const float*)d_in[0];
    const float* W1 = (const float*)d_in[1];
    const float* b1 = (const float*)d_in[2];
    const float* W2 = (const float*)d_in[3];
    const float* b2 = (const float*)d_in[4];
    const float* W3 = (const float*)d_in[5];
    const float* b3 = (const float*)d_in[6];

    float* out_res = (float*)d_out;
    float* out_lad = out_res + (size_t)NTOT * DCH;

    // zero the log-abs-det accumulator region (atomicAdd target) every call
    zero_lad_kernel<<<(NTOT + 255) / 256, 256, 0, stream>>>(out_lad, NTOT);

    np_prior_kernel<<<DCH * NTILES, 256, 0, stream>>>(
        x, W1, b1, W2, b2, W3, b3, out_res, out_lad);
}
